// Attention_50173807952647
// MI455X (gfx1250) — matrix operand order repaired and checked by reading, not compiled, not run on hardware
//
#include <hip/hip_runtime.h>

typedef __attribute__((ext_vector_type(16))) _Float16 v16h;
typedef __attribute__((ext_vector_type(8)))  _Float16 v8h;
typedef __attribute__((ext_vector_type(8)))  float    v8f;

#define HEADS     16
#define KV_HEADS  8
#define DIM_HEAD  64
#define ROT_DIM   32
#define SEQ       2048
#define BATCH     4
#define DIM       1024
#define KV_DIM    512
#define QKV_COLS  2048
#define TOKENS    (BATCH*SEQ)
#define ATT_SCALE 0.125f   // 64^-0.5

// ---------------------------------------------------------------------------
// Fragment loaders (wave32), using this code's contiguous k order (not the ISA table order).
// 16-bit A (16x32): lane L: M = L%16; VGPR0-3 = K 16*(L/16)+0..7 (packed pairs)
//                                    VGPR4-7 = K 16*(L/16)+8..15
__device__ __forceinline__ v16h load_a_frag(const _Float16* __restrict__ row,
                                            int kbase, int half) {
  v8h lo = *(const v8h*)(row + kbase + 16 * half);
  v8h hi = *(const v8h*)(row + kbase + 16 * half + 8);
  v16h a;
#pragma unroll
  for (int i = 0; i < 8; ++i) { a[i] = lo[i]; a[i + 8] = hi[i]; }
  return a;
}
// 16-bit B (32x16): lane L: N = L%16; VGPR j holds K = 16*(L/16) + 2j, 2j+1
// -> one contiguous v16h from the (transposed) B row.
__device__ __forceinline__ v16h load_b_frag(const _Float16* __restrict__ colrow,
                                            int kbase, int half) {
  return *(const v16h*)(colrow + kbase + 16 * half);
}

__device__ __forceinline__ v8f wmma_f16(v16h a, v16h b, v8f c) {
  return __builtin_amdgcn_wmma_f32_16x16x32_f16(false, a, false, b,
                                                (short)0, c, false, false);
}

// CDNA5 async DMA: copy 16 bytes global -> LDS, tracked by ASYNCcnt.
// VDST operand = LDS byte address (low 32 bits of a generic shared pointer).
__device__ __forceinline__ void async_ld16(void* lds_ptr, const _Float16* g) {
  unsigned int lds_off = (unsigned int)(unsigned long long)lds_ptr;
  asm volatile("global_load_async_to_lds_b128 %0, %1, off"
               :: "v"(lds_off), "v"(g) : "memory");
}

// ---------------------------------------------------------------------------
// Kernel 1a: f32 -> f16 elementwise
__global__ void cvt_f16(const float* __restrict__ src, _Float16* __restrict__ dst,
                        int n) {
  int i = blockIdx.x * blockDim.x + threadIdx.x;
  if (i < n) dst[i] = (_Float16)src[i];
}

// Kernel 1b: transpose-convert W[K x N] (row-major) -> Wt[N x K] f16
__global__ void transpose_f16(const float* __restrict__ src,
                              _Float16* __restrict__ dst, int K, int N) {
  int i = blockIdx.x * blockDim.x + threadIdx.x;
  if (i >= K * N) return;
  int n = i / K, k = i % K;
  dst[(size_t)n * K + k] = (_Float16)src[(size_t)k * N + n];
}

// ---------------------------------------------------------------------------
// Kernel 2/5: GEMM  C[M x ldc] = A[M x K]_f16 @ Bt[N x K]_f16 (+bias), f32 out.
// Block = 128 threads (4 waves); each wave computes 32 rows x 64 cols
// (2 M-subtiles x 4 N-tiles = 8 WMMAs per K-step of 32; B frags reused).
__global__ void gemm_f16_wmma(const _Float16* __restrict__ A,
                              const _Float16* __restrict__ Bt,
                              float* __restrict__ C,
                              const float* __restrict__ bias,
                              int K, int ldc) {
  const int lane = threadIdx.x & 31;
  const int wave = threadIdx.x >> 5;
  const int l16  = lane & 15;
  const int half = lane >> 4;
  const int m0   = blockIdx.x * 32;
  const int n0   = (blockIdx.y * 4 + wave) * 64;

  const _Float16* arow0 = A + (size_t)(m0 + l16) * K;
  const _Float16* arow1 = A + (size_t)(m0 + 16 + l16) * K;
  v8f acc[2][4] = {};
  for (int k = 0; k < K; k += 32) {
    v16h a0 = load_a_frag(arow0, k, half);
    v16h a1 = load_a_frag(arow1, k, half);
#pragma unroll
    for (int nt = 0; nt < 4; ++nt) {
      const _Float16* bcol = Bt + (size_t)(n0 + nt * 16 + l16) * K;
      v16h bf = load_b_frag(bcol, k, half);
      acc[0][nt] = wmma_f16(a0, bf, acc[0][nt]);
      acc[1][nt] = wmma_f16(a1, bf, acc[1][nt]);
    }
  }
  // C layout: VGPR r -> row r + 8*(lane/16); lane%16 -> column
#pragma unroll
  for (int ms = 0; ms < 2; ++ms) {
#pragma unroll
    for (int nt = 0; nt < 4; ++nt) {
#pragma unroll
      for (int r = 0; r < 8; ++r) {
        int row = m0 + ms * 16 + r + 8 * half;
        int col = n0 + nt * 16 + l16;
        float b = bias ? bias[col] : 0.f;
        C[(size_t)row * ldc + col] = acc[ms][nt][r] + b;
      }
    }
  }
}

// ---------------------------------------------------------------------------
// Kernel 3: interleaved RoPE on first ROT_DIM dims of each Q/K head, then
// scatter into per-head f16 buffers: Qh[b,H,n,d], Kh[b,Hk,n,d], Vt[b,Hk,d,n].
__global__ void rope_scatter(const float* __restrict__ QKV,
                             _Float16* __restrict__ Qh,
                             _Float16* __restrict__ Kh,
                             _Float16* __restrict__ Vt) {
  const int row = blockIdx.x;            // token row: b*SEQ + t
  const int b = row / SEQ, t = row % SEQ;
  const int c0 = threadIdx.x * 8;        // 8 contiguous cols, within one head
  float v[8];
#pragma unroll
  for (int j = 0; j < 8; ++j) v[j] = QKV[(size_t)row * QKV_COLS + c0 + j];

  const int d0  = c0 & (DIM_HEAD - 1);
  const bool isQ = c0 < DIM;
  const bool isK = (c0 >= DIM) && (c0 < DIM + KV_DIM);
  if ((isQ || isK) && d0 < ROT_DIM) {
#pragma unroll
    for (int j = 0; j < 4; ++j) {
      int i = (d0 >> 1) + j;                       // frequency index
      float inv = __expf(-(2.f * (float)i / (float)ROT_DIM) * 9.210340372f);
      float ang = (float)t * inv;
      float cs = __cosf(ang), sn = __sinf(ang);
      float x0 = v[2 * j], x1 = v[2 * j + 1];
      v[2 * j]     = x0 * cs - x1 * sn;
      v[2 * j + 1] = x1 * cs + x0 * sn;
    }
  }
  if (isQ) {
    int h = c0 >> 6;
    _Float16* dst = Qh + (((size_t)(b * HEADS + h) * SEQ + t) * DIM_HEAD + d0);
#pragma unroll
    for (int j = 0; j < 8; ++j) dst[j] = (_Float16)v[j];
  } else if (isK) {
    int h = (c0 - DIM) >> 6;
    _Float16* dst = Kh + (((size_t)(b * KV_HEADS + h) * SEQ + t) * DIM_HEAD + d0);
#pragma unroll
    for (int j = 0; j < 8; ++j) dst[j] = (_Float16)v[j];
  } else {
    int h = (c0 - DIM - KV_DIM) >> 6;
    _Float16* base = Vt + ((size_t)(b * KV_HEADS + h) * DIM_HEAD) * SEQ;
#pragma unroll
    for (int j = 0; j < 8; ++j)
      base[(size_t)(d0 + j) * SEQ + t] = (_Float16)v[j];
  }
}

// ---------------------------------------------------------------------------
// Kernel 4: flash attention. Grid (SEQ/64, B*H); block 128 (4 waves).
// Wave handles 16 q rows; KV tiles of 32 tokens staged block-wide into LDS
// via CDNA5 async-to-LDS DMA (all 4 waves share the same K/V tile).
__global__ void flash_attn(const _Float16* __restrict__ Qh,
                           const _Float16* __restrict__ Kh,
                           const _Float16* __restrict__ Vt,
                           _Float16* __restrict__ Oh) {
  __shared__ __align__(16) _Float16 ktile[32 * DIM_HEAD];  // [token][d]   4 KB
  __shared__ __align__(16) _Float16 vtile[DIM_HEAD * 32];  // [d][token]   4 KB
  __shared__ __align__(32) _Float16 pbuf[4][16 * 32];      // per-wave P   4 KB

  const int tid  = threadIdx.x;
  const int lane = tid & 31;
  const int wave = tid >> 5;
  const int l16  = lane & 15;
  const int half = lane >> 4;
  const int bh = blockIdx.y;             // 0..B*H-1
  const int b = bh / HEADS, h = bh % HEADS;
  const int hk = h >> 1;                 // repeat_interleave GQA mapping
  const int q0 = blockIdx.x * 64 + wave * 16;

  const _Float16* Qbase = Qh + ((size_t)bh * SEQ + q0 + l16) * DIM_HEAD;
  v16h aQ0 = load_a_frag(Qbase, 0, half);
  v16h aQ1 = load_a_frag(Qbase, 32, half);

  const _Float16* Kb = Kh + (size_t)(b * KV_HEADS + hk) * SEQ * DIM_HEAD;
  const _Float16* Vb = Vt + (size_t)(b * KV_HEADS + hk) * DIM_HEAD * SEQ;

  v8f oacc[4] = {};
  float m[8], l[8];
#pragma unroll
  for (int r = 0; r < 8; ++r) { m[r] = -1e30f; l[r] = 0.f; }

  _Float16* pw = &pbuf[wave][0];

  for (int kv = 0; kv < SEQ; kv += 32) {
    // --- block-cooperative async DMA of K (32x64) and V (64x32) into LDS ---
    // 256 16-byte chunks each; 2 per thread per array.
#pragma unroll
    for (int s = 0; s < 2; ++s) {
      int c = tid + s * 128;                       // 0..255
      int krow = c >> 3, kseg = c & 7;             // K: 8 chunks per 64-h row
      async_ld16(&ktile[c * 8],
                 Kb + (size_t)(kv + krow) * DIM_HEAD + kseg * 8);
      int vrow = c >> 2, vseg = c & 3;             // V: 4 chunks per 32-h row
      async_ld16(&vtile[c * 8],
                 Vb + (size_t)vrow * SEQ + kv + vseg * 8);
    }
    asm volatile("s_wait_asynccnt 0x0" ::: "memory");
    __syncthreads();

    // S = Q K^T : two 16x16 column tiles, K-dim (d=64) split into two x32
    v8f s0 = {}, s1 = {};
#pragma unroll
    for (int ds = 0; ds < 2; ++ds) {
      v16h bk0 = load_b_frag(&ktile[(l16)      * DIM_HEAD], ds * 32, half);
      v16h bk1 = load_b_frag(&ktile[(16 + l16) * DIM_HEAD], ds * 32, half);
      v16h aq  = (ds == 0) ? aQ0 : aQ1;
      s0 = wmma_f16(aq, bk0, s0);
      s1 = wmma_f16(aq, bk1, s1);
    }

    // online softmax update (rows striped: VGPR r -> row r+8*half, lane%16 -> col)
#pragma unroll
    for (int r = 0; r < 8; ++r) {
      float v0 = s0[r] * ATT_SCALE;
      float v1 = s1[r] * ATT_SCALE;
      float mx = fmaxf(v0, v1);
#pragma unroll
      for (int off = 1; off < 16; off <<= 1)
        mx = fmaxf(mx, __shfl_xor(mx, off, 32));
      float mnew  = fmaxf(m[r], mx);
      float alpha = __expf(m[r] - mnew);
      float p0 = __expf(v0 - mnew);
      float p1 = __expf(v1 - mnew);
      float ps = p0 + p1;
#pragma unroll
      for (int off = 1; off < 16; off <<= 1)
        ps += __shfl_xor(ps, off, 32);
      l[r] = l[r] * alpha + ps;
      m[r] = mnew;
#pragma unroll
      for (int dt = 0; dt < 4; ++dt) oacc[dt][r] *= alpha;
      int prow = r + 8 * half;
      pw[prow * 32 + l16]      = (_Float16)p0;
      pw[prow * 32 + 16 + l16] = (_Float16)p1;
    }
    asm volatile("s_wait_dscnt 0x0" ::: "memory");

    // reload P (16x32) as A fragment from LDS, then O += P V (V from LDS)
    v16h aP = load_a_frag(pw + l16 * 32, 0, half);
#pragma unroll
    for (int dt = 0; dt < 4; ++dt) {
      v16h bV = *(const v16h*)(&vtile[(dt * 16 + l16) * 32 + 16 * half]);
      oacc[dt] = wmma_f16(aP, bV, oacc[dt]);
    }
    __syncthreads();   // protect ktile/vtile from next iteration's DMA (WAR)
  }

  // normalize and store O as [b, n, H*dh] f16 for the output projection
#pragma unroll
  for (int r = 0; r < 8; ++r) {
    float inv = 1.f / l[r];
    int q = q0 + r + 8 * half;
#pragma unroll
    for (int dt = 0; dt < 4; ++dt) {
      float o = oacc[dt][r] * inv;
      Oh[((size_t)(b * SEQ + q) * DIM) + h * DIM_HEAD + dt * 16 + l16] =
          (_Float16)o;
    }
  }
}

// ---------------------------------------------------------------------------
extern "C" void kernel_launch(void* const* d_in, const int* in_sizes, int n_in,
                              void* d_out, int out_size, void* d_ws, size_t ws_size,
                              hipStream_t stream) {
  const float* x  = (const float*)d_in[0];   // [4,2048,1024]
  const float* Wq = (const float*)d_in[1];   // [1024,1024]
  const float* Wk = (const float*)d_in[2];   // [1024,512]
  const float* Wv = (const float*)d_in[3];   // [1024,512]
  const float* Wo = (const float*)d_in[4];   // [1024,1024]
  const float* bo = (const float*)d_in[5];   // [1024]
  float* out = (float*)d_out;                // [4,2048,1024]

  char* ws = (char*)d_ws;
  size_t off = 0;
  _Float16* xh    = (_Float16*)(ws + off); off += (size_t)TOKENS * DIM * 2;      // 16 MB
  _Float16* WqkvT = (_Float16*)(ws + off); off += (size_t)QKV_COLS * DIM * 2;    //  4 MB
  _Float16* WoT   = (_Float16*)(ws + off); off += (size_t)DIM * DIM * 2;         //  2 MB
  float*    QKV   = (float*)   (ws + off); off += (size_t)TOKENS * QKV_COLS * 4; // 64 MB
  _Float16* Qh    = (_Float16*)(ws + off); off += (size_t)BATCH * HEADS * SEQ * DIM_HEAD * 2;
  _Float16* Kh    = (_Float16*)(ws + off); off += (size_t)BATCH * KV_HEADS * SEQ * DIM_HEAD * 2;
  _Float16* Vt    = (_Float16*)(ws + off); off += (size_t)BATCH * KV_HEADS * SEQ * DIM_HEAD * 2;
  _Float16* Oh    = (_Float16*)(ws + off); off += (size_t)TOKENS * DIM * 2;

  // 1) convert x, transpose-convert weights
  {
    int n = TOKENS * DIM;
    cvt_f16<<<(n + 255) / 256, 256, 0, stream>>>(x, xh, n);
  }
  transpose_f16<<<(DIM * DIM + 255) / 256, 256, 0, stream>>>(Wq, WqkvT, DIM, DIM);
  transpose_f16<<<(DIM * KV_DIM + 255) / 256, 256, 0, stream>>>(
      Wk, WqkvT + (size_t)DIM * DIM, DIM, KV_DIM);
  transpose_f16<<<(DIM * KV_DIM + 255) / 256, 256, 0, stream>>>(
      Wv, WqkvT + (size_t)(DIM + KV_DIM) * DIM, DIM, KV_DIM);
  transpose_f16<<<(DIM * DIM + 255) / 256, 256, 0, stream>>>(Wo, WoT, DIM, DIM);

  // 2) fused QKV projection: [8192 x 1024] @ [1024 x 2048] -> f32
  gemm_f16_wmma<<<dim3(TOKENS / 32, QKV_COLS / 256), 128, 0, stream>>>(
      xh, WqkvT, QKV, nullptr, DIM, QKV_COLS);

  // 3) RoPE + scatter into per-head layouts (V transposed)
  rope_scatter<<<TOKENS, 256, 0, stream>>>(QKV, Qh, Kh, Vt);

  // 4) flash attention (async-to-LDS staged K/V tiles)
  flash_attn<<<dim3(SEQ / 64, BATCH * HEADS), 128, 0, stream>>>(Qh, Kh, Vt, Oh);

  // 5) output projection + bias: [8192 x 1024] @ [1024 x 1024] + bo -> f32 out
  gemm_f16_wmma<<<dim3(TOKENS / 32, DIM / 256), 128, 0, stream>>>(
      Oh, WoT, out, bo, DIM, DIM);
}
